// CompGCN_81114752352452
// MI455X (gfx1250) — compile-verified, matrix operand order checked by reading, not run on hardware
//
#include <hip/hip_runtime.h>
#include <hip/hip_bf16.h>

typedef unsigned int   u32;
typedef unsigned short u16;
typedef __attribute__((ext_vector_type(16))) __bf16 v16bf;
typedef __attribute__((ext_vector_type(8)))  float  v8f;

#define DCH 128        // feature dim
#define SA  130        // LDS row stride in u16 (260B = 65 dwords -> conflict-free)
#define BN_EPS 1e-5f

union FragB { u32 u[8]; v16bf v; };

// Manual f32 -> bf16 RNE (used only in the one-time weight convert kernel).
__device__ __forceinline__ u32 f2bf(float f) {
  u32 u = __float_as_uint(f);
  return ((u + 0x7FFFu + ((u >> 16) & 1u)) >> 16) & 0xFFFFu;
}

// Packed f32x2 -> bf16x2 (RNE); a in low 16 bits, b in high 16 bits.
__device__ __forceinline__ u32 pack2bf(float a, float b) {
  union { __hip_bfloat162 h; u32 u; } c;
  c.h = __float22bfloat162_rn(make_float2(a, b));
  return c.u;
}

// Load one 16x32 bf16 fragment (A-layout) from an LDS row.
// Per ISA 7.12.2: lane half h, VGPR v -> K = 32*kt + (v>=4?16:0) + h*8 + (v&3)*2
// => uint index within row = kt*16 + hi*4 + {0..3} and + 8 + {0..3}
__device__ __forceinline__ void load_frag(const u16* rowBase, int kt, int hi, FragB& f) {
  const u32* p = (const u32*)rowBase + kt * 16 + hi * 4;
#pragma unroll
  for (int v = 0; v < 4; ++v) f.u[v] = p[v];
#pragma unroll
  for (int v = 0; v < 4; ++v) f.u[4 + v] = p[8 + v];
}

// ---------------------------------------------------------------------------
// Edge message GEMM + scatter:  agg[dst] += norm * ((x[src]*r[etype]) @ W)
// One block = 128 edges. First-half tiles use WtIn, second half WtOut
// (E/2 is a multiple of 128 so a tile never straddles the split).
// ---------------------------------------------------------------------------
__global__ __launch_bounds__(256) void edge_gemm_scatter(
    const float* __restrict__ x, const float* __restrict__ r,
    const u16* __restrict__ WtIn, const u16* __restrict__ WtOut,
    const int* __restrict__ src, const int* __restrict__ dst,
    const int* __restrict__ etype, const float* __restrict__ enorm,
    float* __restrict__ agg, int halfE)
{
  extern __shared__ char smem[];
  u16*   lA    = (u16*)smem;                       // [128][SA] composed edge rows (bf16)
  u16*   lB    = lA + DCH * SA;                    // [128][SA] W^T rows (bf16)
  float* lNorm = (float*)(lB + DCH * SA);          // [128]
  int*   lDst  = (int*)(lNorm + DCH);              // [128]

  const int tid      = threadIdx.x;
  const int tileBase = blockIdx.x * 128;
  const u16* Wt = (tileBase < halfE) ? WtIn : WtOut;

  // Stage W^T (row n holds W[:,n], contiguous in k) -> lB
  {
    int row = tid >> 1, half = tid & 1;
    const u32* g = (const u32*)(Wt + row * DCH) + half * 32;
    u32* s = (u32*)(lB + row * SA) + half * 32;
#pragma unroll
    for (int i = 0; i < 32; i += 4) {
      uint4 v4 = ((const uint4*)g)[i >> 2];
      s[i] = v4.x; s[i + 1] = v4.y; s[i + 2] = v4.z; s[i + 3] = v4.w;
    }
  }
  // Gather + compose edge rows: lA[e][d] = bf16(x[src][d] * r[etype][d])
  {
    int le = tid >> 1, half = tid & 1;
    int e = tileBase + le;
    int s_ = src[e], t_ = etype[e];
    if (half == 0) { lNorm[le] = enorm[e]; lDst[le] = dst[e]; }
    const float4* xv = (const float4*)(x + (size_t)s_ * DCH + half * 64);
    const float4* rv = (const float4*)(r + (size_t)t_ * DCH + half * 64);
    u32* arow = (u32*)(lA + le * SA) + half * 32;
#pragma unroll
    for (int i = 0; i < 16; ++i) {
      float4 a = xv[i], b = rv[i];
      arow[2 * i]     = pack2bf(a.x * b.x, a.y * b.y);
      arow[2 * i + 1] = pack2bf(a.z * b.z, a.w * b.w);
    }
  }
  __syncthreads();

  const int wave = tid >> 5, lane = tid & 31;
  const int lrow = lane & 15, hi = lane >> 4;
  const u16* aRow = lA + (wave * 16 + lrow) * SA;

  // Hoist per-accumulator-row scatter pointers & norms (C layout:
  // lane,VGPR v -> M = v + 8*hi, N = lane&15).
  float* aggP[8];
  float  nrm[8];
#pragma unroll
  for (int v = 0; v < 8; ++v) {
    int m = wave * 16 + v + 8 * hi;
    aggP[v] = agg + (size_t)lDst[m] * DCH + lrow;
    nrm[v]  = lNorm[m];
  }

#pragma unroll
  for (int nt = 0; nt < 8; ++nt) {
    v8f acc = {};
    const u16* bRow = lB + (nt * 16 + lrow) * SA;
#pragma unroll
    for (int kt = 0; kt < 4; ++kt) {
      FragB A, B;
      load_frag(aRow, kt, hi, A);
      load_frag(bRow, kt, hi, B);
      acc = __builtin_amdgcn_wmma_f32_16x16x32_bf16(false, A.v, false, B.v,
                                                    (short)0, acc, false, false);
    }
#pragma unroll
    for (int v = 0; v < 8; ++v)
      atomicAdd(aggP[v] + nt * 16, acc[v] * nrm[v]);   // col offset folds to imm
  }
}

// ---------------------------------------------------------------------------
// Node update: out = tanh( (agg + ((x*loop_rel)@loop_w)/3 + bias) * bnS + bnB )
// One block = 128 nodes (last tile partial).
// ---------------------------------------------------------------------------
__global__ __launch_bounds__(256) void node_update(
    const float* __restrict__ x, const float* __restrict__ loop_rel,
    const u16* __restrict__ WtLoop, const float* __restrict__ agg,
    const float* __restrict__ bias, const float* __restrict__ gamma,
    const float* __restrict__ beta, float* __restrict__ out, int nNodes)
{
  extern __shared__ char smem[];
  u16*   lA  = (u16*)smem;
  u16*   lB  = lA + DCH * SA;
  float* lBi = (float*)(lB + DCH * SA);   // [128] bias
  float* lSc = lBi + DCH;                 // [128] gamma/sqrt(1+eps)
  float* lBe = lSc + DCH;                 // [128] beta

  const int tid  = threadIdx.x;
  const int base = blockIdx.x * 128;

  { // stage loop_w^T
    int row = tid >> 1, half = tid & 1;
    const u32* g = (const u32*)(WtLoop + row * DCH) + half * 32;
    u32* s = (u32*)(lB + row * SA) + half * 32;
#pragma unroll
    for (int i = 0; i < 32; i += 4) {
      uint4 v4 = ((const uint4*)g)[i >> 2];
      s[i] = v4.x; s[i + 1] = v4.y; s[i + 2] = v4.z; s[i + 3] = v4.w;
    }
  }
  if (tid < DCH) {
    lBi[tid] = bias[tid];
    lSc[tid] = gamma[tid] * rsqrtf(1.0f + BN_EPS);
    lBe[tid] = beta[tid];
  }
  { // stage (x * loop_rel) as bf16
    int ln = tid >> 1, half = tid & 1;
    int node = base + ln;
    u32* arow = (u32*)(lA + ln * SA) + half * 32;
    if (node < nNodes) {
      const float4* xv = (const float4*)(x + (size_t)node * DCH + half * 64);
      const float4* rv = (const float4*)(loop_rel + half * 64);
#pragma unroll
      for (int i = 0; i < 16; ++i) {
        float4 a = xv[i], b = rv[i];
        arow[2 * i]     = pack2bf(a.x * b.x, a.y * b.y);
        arow[2 * i + 1] = pack2bf(a.z * b.z, a.w * b.w);
      }
    } else {
#pragma unroll
      for (int i = 0; i < 32; ++i) arow[i] = 0u;
    }
  }
  __syncthreads();

  const int wave = tid >> 5, lane = tid & 31;
  const int lrow = lane & 15, hi = lane >> 4;
  const u16* aRow = lA + (wave * 16 + lrow) * SA;

  // Hoist row pointers per accumulator row.
  const float* aggP[8];
  float*       outP[8];
  bool         ok[8];
#pragma unroll
  for (int v = 0; v < 8; ++v) {
    int m = wave * 16 + v + 8 * hi;
    int node = base + m;
    ok[v]   = (node < nNodes);
    aggP[v] = agg + (size_t)node * DCH + lrow;
    outP[v] = out + (size_t)node * DCH + lrow;
  }

#pragma unroll
  for (int nt = 0; nt < 8; ++nt) {
    v8f acc = {};
    const u16* bRow = lB + (nt * 16 + lrow) * SA;
#pragma unroll
    for (int kt = 0; kt < 4; ++kt) {
      FragB A, B;
      load_frag(aRow, kt, hi, A);
      load_frag(bRow, kt, hi, B);
      acc = __builtin_amdgcn_wmma_f32_16x16x32_bf16(false, A.v, false, B.v,
                                                    (short)0, acc, false, false);
    }
#pragma unroll
    for (int v = 0; v < 8; ++v) {
      if (ok[v]) {
        int col = nt * 16 + lrow;
        float h = aggP[v][nt * 16] + acc[v] * (1.0f / 3.0f) + lBi[col];
        h = h * lSc[col] + lBe[col];
        outP[v][nt * 16] = tanhf(h);
      }
    }
  }
}

// ---------------------------------------------------------------------------
// Relation update: r_next(32x128) = r(32x128) @ w_rel(128x128). One block.
// ---------------------------------------------------------------------------
__global__ __launch_bounds__(256) void rel_update(
    const float* __restrict__ r, const u16* __restrict__ WtRel,
    float* __restrict__ rNext)
{
  extern __shared__ char smem[];
  u16* lA = (u16*)smem;          // [32][SA]
  u16* lB = lA + 32 * SA;        // [128][SA]

  const int tid = threadIdx.x;
  { // stage w_rel^T
    int row = tid >> 1, half = tid & 1;
    const u32* g = (const u32*)(WtRel + row * DCH) + half * 32;
    u32* s = (u32*)(lB + row * SA) + half * 32;
#pragma unroll
    for (int i = 0; i < 32; i += 4) {
      uint4 v4 = ((const uint4*)g)[i >> 2];
      s[i] = v4.x; s[i + 1] = v4.y; s[i + 2] = v4.z; s[i + 3] = v4.w;
    }
  }
  if (tid < 64) { // stage r as bf16
    int row = tid >> 1, half = tid & 1;
    const float4* rv = (const float4*)(r + (size_t)row * DCH + half * 64);
    u32* arow = (u32*)(lA + row * SA) + half * 32;
#pragma unroll
    for (int i = 0; i < 16; ++i) {
      float4 a = rv[i];
      arow[2 * i]     = pack2bf(a.x, a.y);
      arow[2 * i + 1] = pack2bf(a.z, a.w);
    }
  }
  __syncthreads();

  const int wave = tid >> 5, lane = tid & 31;
  const int lrow = lane & 15, hi = lane >> 4;

#pragma unroll
  for (int tt = 0; tt < 2; ++tt) {
    int tile = wave * 2 + tt;           // 16 output tiles total
    int mt = tile >> 3, nt = tile & 7;
    const u16* aRow = lA + (mt * 16 + lrow) * SA;
    const u16* bRow = lB + (nt * 16 + lrow) * SA;
    v8f acc = {};
#pragma unroll
    for (int kt = 0; kt < 4; ++kt) {
      FragB A, B;
      load_frag(aRow, kt, hi, A);
      load_frag(bRow, kt, hi, B);
      acc = __builtin_amdgcn_wmma_f32_16x16x32_bf16(false, A.v, false, B.v,
                                                    (short)0, acc, false, false);
    }
#pragma unroll
    for (int v = 0; v < 8; ++v) {
      int m = mt * 16 + v + 8 * hi;
      rNext[(size_t)m * DCH + nt * 16 + lrow] = acc[v];
    }
  }
}

// ------------------------------ helpers ------------------------------------
__global__ void zero_f32(float* p, int n) {
  int i = blockIdx.x * blockDim.x + threadIdx.x;
  if (i < n) p[i] = 0.0f;
}
__global__ void deg_kernel(const int* __restrict__ dst, float* __restrict__ deg, int e) {
  int i = blockIdx.x * blockDim.x + threadIdx.x;
  if (i < e) atomicAdd(&deg[dst[i]], 1.0f);
}
__global__ void norm_kernel(float* deg, int n) {
  int i = blockIdx.x * blockDim.x + threadIdx.x;
  if (i < n) { float d = deg[i]; deg[i] = (d > 0.0f) ? rsqrtf(d) : 0.0f; }
}
__global__ void enorm_kernel(const int* __restrict__ src, const int* __restrict__ dst,
                             const float* __restrict__ nrm, float* __restrict__ en, int e) {
  int i = blockIdx.x * blockDim.x + threadIdx.x;
  if (i < e) en[i] = nrm[dst[i]] * nrm[src[i]];
}
// W (128x128 f32, row-major [k][n]) -> Wt (bf16, row-major [n][k])
__global__ void cvt_t(const float* __restrict__ W, u16* __restrict__ Wt) {
  int i = blockIdx.x * blockDim.x + threadIdx.x;
  if (i < DCH * DCH) {
    int k = i / DCH, n = i % DCH;
    Wt[(size_t)n * DCH + k] = (u16)f2bf(W[i]);
  }
}

// ---------------------------------------------------------------------------
extern "C" void kernel_launch(void* const* d_in, const int* in_sizes, int n_in,
                              void* d_out, int out_size, void* d_ws, size_t ws_size,
                              hipStream_t stream) {
  const float* x0       = (const float*)d_in[0];
  const float* init_rel = (const float*)d_in[1];
  const float* in_w     = (const float*)d_in[2];
  const float* out_w    = (const float*)d_in[3];
  const float* loop_w   = (const float*)d_in[4];
  const float* w_rel    = (const float*)d_in[5];
  const float* loop_rel = (const float*)d_in[6];
  const float* bias     = (const float*)d_in[7];
  const float* gamma    = (const float*)d_in[8];
  const float* beta     = (const float*)d_in[9];
  const int*   src      = (const int*)d_in[10];
  const int*   dst      = (const int*)d_in[11];
  const int*   etype    = (const int*)d_in[12];

  const int D   = DCH;
  const int Nn  = in_sizes[0] / D;
  const int Ee  = in_sizes[10];
  const int Ln  = in_sizes[7] / D;     // layers
  const int MAT = D * D;

  // workspace layout
  char* w = (char*)d_ws;
  size_t off = 0;
  auto alloc = [&](size_t bytes) -> char* {
    char* p = w + off;
    off += (bytes + 255) & ~(size_t)255;
    return p;
  };
  float* agg   = (float*)alloc((size_t)Nn * D * 4);
  float* xm0   = (float*)alloc((size_t)Nn * D * 4);
  float* xm1   = (float*)alloc((size_t)Nn * D * 4);
  float* enorm = (float*)alloc((size_t)Ee * 4);
  float* deg   = (float*)alloc((size_t)Nn * 4);
  float* rb0   = (float*)alloc((size_t)32 * D * 4);
  float* rb1   = (float*)alloc((size_t)32 * D * 4);
  u16* WtIn    = (u16*)alloc((size_t)Ln * MAT * 2);
  u16* WtOut   = (u16*)alloc((size_t)Ln * MAT * 2);
  u16* WtLoop  = (u16*)alloc((size_t)Ln * MAT * 2);
  u16* WtRel   = (u16*)alloc((size_t)Ln * MAT * 2);
  (void)ws_size; (void)n_in; (void)out_size;

  // weight convert + transpose to bf16
  int cb = (MAT + 255) / 256;
  for (int i = 0; i < Ln; ++i) {
    cvt_t<<<cb, 256, 0, stream>>>(in_w   + (size_t)i * MAT, WtIn   + (size_t)i * MAT);
    cvt_t<<<cb, 256, 0, stream>>>(out_w  + (size_t)i * MAT, WtOut  + (size_t)i * MAT);
    cvt_t<<<cb, 256, 0, stream>>>(loop_w + (size_t)i * MAT, WtLoop + (size_t)i * MAT);
    cvt_t<<<cb, 256, 0, stream>>>(w_rel  + (size_t)i * MAT, WtRel  + (size_t)i * MAT);
  }

  // degree -> norm -> edge_norm
  int nb = (Nn + 255) / 256, eb = (Ee + 255) / 256;
  zero_f32<<<nb, 256, 0, stream>>>(deg, Nn);
  deg_kernel<<<eb, 256, 0, stream>>>(dst, deg, Ee);
  norm_kernel<<<nb, 256, 0, stream>>>(deg, Nn);
  enorm_kernel<<<eb, 256, 0, stream>>>(src, dst, deg, enorm, Ee);

  const size_t smemE = (size_t)2 * DCH * SA * 2 + DCH * 4 + DCH * 4;
  const size_t smemN = (size_t)2 * DCH * SA * 2 + 3 * DCH * 4;
  const size_t smemR = (size_t)32 * SA * 2 + (size_t)DCH * SA * 2;

  const float* xcur = x0;
  const float* rcur = init_rel;
  float* rnext = rb0;
  for (int i = 0; i < Ln; ++i) {
    zero_f32<<<(Nn * D + 255) / 256, 256, 0, stream>>>(agg, Nn * D);
    edge_gemm_scatter<<<Ee / 128, 256, smemE, stream>>>(
        xcur, rcur, WtIn + (size_t)i * MAT, WtOut + (size_t)i * MAT,
        src, dst, etype, enorm, agg, Ee / 2);
    float* xnext = (i == Ln - 1) ? (float*)d_out : ((i & 1) ? xm1 : xm0);
    node_update<<<(Nn + 127) / 128, 256, smemN, stream>>>(
        xcur, loop_rel + (size_t)i * D, WtLoop + (size_t)i * MAT, agg,
        bias + (size_t)i * D, gamma + (size_t)i * D, beta + (size_t)i * D,
        xnext, Nn);
    if (i < Ln - 1) {
      rel_update<<<1, 256, smemR, stream>>>(rcur, WtRel + (size_t)i * MAT, rnext);
      rcur = rnext;
      rnext = (rnext == rb0) ? rb1 : rb0;
    }
    xcur = xnext;
  }
}
